// FuseFeatureBlock_19799799234891
// MI455X (gfx1250) — matrix operand order repaired and checked by reading, not compiled, not run on hardware
//
#include <hip/hip_runtime.h>
#include <hip/hip_bf16.h>

// ---------------- problem constants ----------------
#define B_    8
#define SQ_   256
#define SKV_  1024
#define D_    1024
#define H_    16
#define DK_   64
#define DFF_  4096

typedef __attribute__((ext_vector_type(16))) __bf16 v16bf;
typedef __attribute__((ext_vector_type(8)))  __bf16 v8bf;
typedef __attribute__((ext_vector_type(8)))  float  v8f;

union BFrag16 { v16bf v; v8bf h[2]; };

static __device__ __forceinline__ __bf16 f2bf(float f) {
    unsigned u = __builtin_bit_cast(unsigned, f);
    unsigned r = (u + 0x7FFFu + ((u >> 16) & 1u)) >> 16;   // round-to-nearest-even
    unsigned short s = (unsigned short)r;
    return __builtin_bit_cast(__bf16, s);
}

// -----------------------------------------------------------------------------------------
// Register-blocked WMMA GEMM core.
// One wave computes an (MT*16) x 64 tile of C = A[M,K](bf16,row-major) * Bt[N,K]^T.
//   A-fragment (16x32, MxK): lane = h*16+m holds K = {kb+8h..kb+8h+7, kb+16+8h..+7} (8/8).
//   B-fragment (32x16, KxN): lane = h*16+n holds K = {kb+8h..kb+8h+7, kb+16+8h..+7} (8/8).
// MT=4 -> 16 WMMA per 16 b128 loads per k-step; all 4 B fragments issued as one clause so
// their latency overlaps the preceding WMMA groups. No LDS, no barriers: B redundancy
// across the 8 waves of a block is served by WGP$/L2 (all bf16 weights are L2-resident).
// -----------------------------------------------------------------------------------------
template<int MT>
static __device__ __forceinline__ void wmma_gemm_tiles(
        const __bf16* __restrict__ A, int lda,
        const __bf16* __restrict__ Bt, int ldb,
        int m0, int n0, int K, v8f acc[MT][4]) {
    const int lane = threadIdx.x & 31;
    const int h    = lane >> 4;
    const int ml   = lane & 15;
#pragma unroll
    for (int i = 0; i < MT; ++i)
#pragma unroll
        for (int j = 0; j < 4; ++j) acc[i][j] = (v8f){};

    for (int kb = 0; kb < K; kb += 32) {
        v16bf a[MT];
#pragma unroll
        for (int i = 0; i < MT; ++i) {
            const __bf16* ap = A + (size_t)(m0 + i * 16 + ml) * lda + kb + 8 * h;
            BFrag16 f;
            f.h[0] = *(const v8bf*)ap;
            f.h[1] = *(const v8bf*)(ap + 16);
            a[i] = f.v;
        }
        v16bf bfr[4];
#pragma unroll
        for (int j = 0; j < 4; ++j) {
            const __bf16* bp = Bt + (size_t)(n0 + j * 16 + ml) * ldb + kb + 8 * h;
            BFrag16 g;
            g.h[0] = *(const v8bf*)bp;
            g.h[1] = *(const v8bf*)(bp + 16);
            bfr[j] = g.v;
        }
        // branch-free in-bounds prefetch of the next A slab
        int kpf = (kb + 32 < K) ? (kb + 32) : kb;
        __builtin_prefetch(A + (size_t)(m0 + ml) * lda + kpf, 0, 3);
#pragma unroll
        for (int j = 0; j < 4; ++j)
#pragma unroll
            for (int i = 0; i < MT; ++i)
                acc[i][j] = __builtin_amdgcn_wmma_f32_16x16x32_bf16(
                    false, a[i], false, bfr[j], (short)0, acc[i][j], false, false);
    }
}

// ---------------- LayerNorm (f32 in -> bf16 out) ----------------
__global__ void ln_bf16_kernel(const float* __restrict__ x, const float* __restrict__ g,
                               const float* __restrict__ bb, __bf16* __restrict__ y, int ncols) {
    const int row = blockIdx.x;
    const float* xr = x + (size_t)row * ncols;
    float s = 0.f, s2 = 0.f;
    for (int c = threadIdx.x; c < ncols; c += 256) { float v = xr[c]; s += v; s2 += v * v; }
    __shared__ float red[512];
    red[threadIdx.x] = s; red[256 + threadIdx.x] = s2;
    __syncthreads();
    for (int o = 128; o > 0; o >>= 1) {
        if (threadIdx.x < o) {
            red[threadIdx.x]       += red[threadIdx.x + o];
            red[256 + threadIdx.x] += red[256 + threadIdx.x + o];
        }
        __syncthreads();
    }
    const float mean = red[0] / (float)ncols;
    const float var  = red[256] / (float)ncols - mean * mean;
    const float rstd = rsqrtf(var + 1e-5f);
    for (int c = threadIdx.x; c < ncols; c += 256) {
        float v = (xr[c] - mean) * rstd * g[c] + bb[c];
        y[(size_t)row * ncols + c] = f2bf(v);
    }
}

// ---------------- weight convert + transpose: W[K][N] f32 -> Wt[N][K] bf16 ----------------
__global__ void wtrans_kernel(const float* __restrict__ W, __bf16* __restrict__ Wt,
                              int K, int N) {
    __shared__ float tile[32][33];
    const int k0 = blockIdx.x * 32, n0 = blockIdx.y * 32;
    const int tx = threadIdx.x & 31, ty = threadIdx.x >> 5;   // 8 rows of 32
    for (int i = ty; i < 32; i += 8) tile[i][tx] = W[(size_t)(k0 + i) * N + n0 + tx];
    __syncthreads();
    for (int i = ty; i < 32; i += 8) Wt[(size_t)(n0 + i) * K + k0 + tx] = f2bf(tile[tx][i]);
}

// ---------------- QKV projection, head split (mode: 0=Q scaled, 1=K, 2=V transposed) ----
template<int MT>
__global__ void __launch_bounds__(256)
qkv_proj_kernel(const __bf16* __restrict__ ln, const __bf16* __restrict__ Wt,
                const float* __restrict__ bias, __bf16* __restrict__ out,
                int logS, int mode, float scale) {
    const int S = 1 << logS;
    const int wave = threadIdx.x >> 5, lane = threadIdx.x & 31;
    const int h = lane >> 4, ml = lane & 15;
    const int m0 = blockIdx.x * (8 * MT * 16) + wave * MT * 16;
    const int n0 = blockIdx.y * 64;
    v8f acc[MT][4];
    wmma_gemm_tiles<MT>(ln, D_, Wt, D_, m0, n0, D_, acc);
#pragma unroll
    for (int j = 0; j < 4; ++j) {
        const int n = n0 + j * 16 + ml;
        const float bv = bias[n];
        const int head = n >> 6, dk = n & 63;
#pragma unroll
        for (int i = 0; i < MT; ++i)
#pragma unroll
            for (int r = 0; r < 8; ++r) {
                int m = m0 + i * 16 + 8 * h + r;
                int b = m >> logS, s = m & (S - 1);
                __bf16 o = f2bf((acc[i][j][r] + bv) * scale);
                if (mode == 2)  // Vt[b][head][dk][s]
                    out[(((size_t)b * H_ + head) * DK_ + dk) * S + s] = o;
                else            // Qh/Kh[b][head][s][dk]
                    out[(((size_t)b * H_ + head) * S + s) * DK_ + dk] = o;
            }
    }
}

// ---------------- scores = Q @ K^T (scale pre-folded into Q), f32 -> attn region --------
template<int MT>
__global__ void __launch_bounds__(256)
scores_kernel(const __bf16* __restrict__ Qh, const __bf16* __restrict__ Kh,
              float* __restrict__ attn) {
    const int bh = blockIdx.z;
    const __bf16* Q = Qh + (size_t)bh * SQ_ * DK_;
    const __bf16* K = Kh + (size_t)bh * SKV_ * DK_;
    const int wave = threadIdx.x >> 5, lane = threadIdx.x & 31;
    const int h = lane >> 4, ml = lane & 15;
    const int m0 = blockIdx.x * (8 * MT * 16) + wave * MT * 16;
    const int n0 = blockIdx.y * 64;
    v8f acc[MT][4];
    wmma_gemm_tiles<MT>(Q, DK_, K, DK_, m0, n0, DK_, acc);
    float* Sr = attn + (size_t)bh * SQ_ * SKV_;
#pragma unroll
    for (int j = 0; j < 4; ++j) {
        const int n = n0 + j * 16 + ml;
#pragma unroll
        for (int i = 0; i < MT; ++i)
#pragma unroll
            for (int r = 0; r < 8; ++r)
                Sr[(size_t)(m0 + i * 16 + 8 * h + r) * SKV_ + n] = acc[i][j][r];
    }
}

// ---------------- row softmax in-place (f32) + bf16 copy for ctx GEMM -------------------
__global__ void softmax_kernel(float* __restrict__ attn, __bf16* __restrict__ P) {
    const int row = blockIdx.x;
    float* a = attn + (size_t)row * SKV_;
    __shared__ float red[256];
    float mx = -3.402823466e38f;
    for (int c = threadIdx.x; c < SKV_; c += 256) mx = fmaxf(mx, a[c]);
    red[threadIdx.x] = mx; __syncthreads();
    for (int o = 128; o > 0; o >>= 1) {
        if (threadIdx.x < o) red[threadIdx.x] = fmaxf(red[threadIdx.x], red[threadIdx.x + o]);
        __syncthreads();
    }
    mx = red[0]; __syncthreads();
    float s = 0.f;
    for (int c = threadIdx.x; c < SKV_; c += 256) { float e = expf(a[c] - mx); a[c] = e; s += e; }
    red[threadIdx.x] = s; __syncthreads();
    for (int o = 128; o > 0; o >>= 1) {
        if (threadIdx.x < o) red[threadIdx.x] += red[threadIdx.x + o];
        __syncthreads();
    }
    const float inv = 1.0f / red[0];
    for (int c = threadIdx.x; c < SKV_; c += 256) {
        float p = a[c] * inv;
        a[c] = p;
        P[(size_t)row * SKV_ + c] = f2bf(p);
    }
}

// ---------------- ctx = P @ V  (Vt is [dk][skv] so it is already B^T), merge heads ------
template<int MT>
__global__ void __launch_bounds__(256)
ctx_kernel(const __bf16* __restrict__ P, const __bf16* __restrict__ Vt,
           __bf16* __restrict__ ctx) {
    const int bh = blockIdx.z;
    const int b = bh >> 4, hd = bh & 15;
    const __bf16* A  = P  + (size_t)bh * SQ_ * SKV_;
    const __bf16* Bt = Vt + (size_t)bh * DK_ * SKV_;
    const int wave = threadIdx.x >> 5, lane = threadIdx.x & 31;
    const int h = lane >> 4, ml = lane & 15;
    const int m0 = blockIdx.x * (8 * MT * 16) + wave * MT * 16;
    v8f acc[MT][4];
    wmma_gemm_tiles<MT>(A, SKV_, Bt, SKV_, m0, 0, SKV_, acc);
#pragma unroll
    for (int j = 0; j < 4; ++j) {
        const int n = j * 16 + ml;
#pragma unroll
        for (int i = 0; i < MT; ++i)
#pragma unroll
            for (int r = 0; r < 8; ++r) {
                int m = m0 + i * 16 + 8 * h + r;
                ctx[((size_t)b * SQ_ + m) * D_ + hd * DK_ + n] = f2bf(acc[i][j][r]);
            }
    }
}

// ---------------- generic GEMM -> f32 (+bias, optional residual) ------------------------
template<int MT>
__global__ void __launch_bounds__(256)
gemm_f32out_kernel(const __bf16* __restrict__ A, const __bf16* __restrict__ Wt,
                   const float* __restrict__ bias, const float* __restrict__ residual,
                   float* __restrict__ out, int Kdim, int Ndim) {
    const int wave = threadIdx.x >> 5, lane = threadIdx.x & 31;
    const int h = lane >> 4, ml = lane & 15;
    const int m0 = blockIdx.x * (8 * MT * 16) + wave * MT * 16;
    const int n0 = blockIdx.y * 64;
    v8f acc[MT][4];
    wmma_gemm_tiles<MT>(A, Kdim, Wt, Kdim, m0, n0, Kdim, acc);
#pragma unroll
    for (int j = 0; j < 4; ++j) {
        const int n = n0 + j * 16 + ml;
        const float bv = bias[n];
#pragma unroll
        for (int i = 0; i < MT; ++i)
#pragma unroll
            for (int r = 0; r < 8; ++r) {
                int m = m0 + i * 16 + 8 * h + r;
                float v = acc[i][j][r] + bv;
                if (residual) v += residual[(size_t)m * Ndim + n];
                out[(size_t)m * Ndim + n] = v;
            }
    }
}

// ---------------- GEMM + erf-GELU -> bf16 -----------------------------------------------
template<int MT>
__global__ void __launch_bounds__(256)
gemm_gelu_kernel(const __bf16* __restrict__ A, const __bf16* __restrict__ Wt,
                 const float* __restrict__ bias, __bf16* __restrict__ out,
                 int Kdim, int Ndim) {
    const int wave = threadIdx.x >> 5, lane = threadIdx.x & 31;
    const int h = lane >> 4, ml = lane & 15;
    const int m0 = blockIdx.x * (8 * MT * 16) + wave * MT * 16;
    const int n0 = blockIdx.y * 64;
    v8f acc[MT][4];
    wmma_gemm_tiles<MT>(A, Kdim, Wt, Kdim, m0, n0, Kdim, acc);
#pragma unroll
    for (int j = 0; j < 4; ++j) {
        const int n = n0 + j * 16 + ml;
        const float bv = bias[n];
#pragma unroll
        for (int i = 0; i < MT; ++i)
#pragma unroll
            for (int r = 0; r < 8; ++r) {
                int m = m0 + i * 16 + 8 * h + r;
                float v = acc[i][j][r] + bv;
                float gl = 0.5f * v * (1.0f + erff(v * 0.70710678118654752f));
                out[(size_t)m * Ndim + n] = f2bf(gl);
            }
    }
}

// =========================================================================================
extern "C" void kernel_launch(void* const* d_in, const int* in_sizes, int n_in,
                              void* d_out, int out_size, void* d_ws, size_t ws_size,
                              hipStream_t stream) {
    const float* memory = (const float*)d_in[0];
    const float* q      = (const float*)d_in[1];
    const float* Wq = (const float*)d_in[2];  const float* bq = (const float*)d_in[3];
    const float* Wk = (const float*)d_in[4];  const float* bk = (const float*)d_in[5];
    const float* Wv = (const float*)d_in[6];  const float* bv = (const float*)d_in[7];
    const float* Wo = (const float*)d_in[8];  const float* bo = (const float*)d_in[9];
    const float* g_kv = (const float*)d_in[10]; const float* b_kv = (const float*)d_in[11];
    const float* g_q  = (const float*)d_in[12]; const float* b_q  = (const float*)d_in[13];
    const float* g_ff = (const float*)d_in[14]; const float* b_ff = (const float*)d_in[15];
    const float* W1 = (const float*)d_in[16]; const float* b1 = (const float*)d_in[17];
    const float* W2 = (const float*)d_in[18]; const float* b2 = (const float*)d_in[19];

    float* out_f32  = (float*)d_out;                                   // [B,SQ,D]
    float* attn_f32 = (float*)d_out + (size_t)B_ * SQ_ * D_;           // [B,H,SQ,SKV]

    // workspace carve-up (256B aligned)
    char* wsb = (char*)d_ws;
    size_t off = 0;
    auto alloc = [&](size_t bytes) -> char* {
        char* p = wsb + off;
        off = (off + bytes + 255) & ~(size_t)255;
        return p;
    };
    __bf16* q_ln  = (__bf16*)alloc((size_t)B_ * SQ_  * D_ * 2);
    __bf16* kv_ln = (__bf16*)alloc((size_t)B_ * SKV_ * D_ * 2);
    __bf16* Wq_t  = (__bf16*)alloc((size_t)D_ * D_ * 2);
    __bf16* Wk_t  = (__bf16*)alloc((size_t)D_ * D_ * 2);
    __bf16* Wv_t  = (__bf16*)alloc((size_t)D_ * D_ * 2);
    __bf16* Wo_t  = (__bf16*)alloc((size_t)D_ * D_ * 2);
    __bf16* W1_t  = (__bf16*)alloc((size_t)D_ * DFF_ * 2);
    __bf16* W2_t  = (__bf16*)alloc((size_t)DFF_ * D_ * 2);
    __bf16* Qh    = (__bf16*)alloc((size_t)B_ * H_ * SQ_  * DK_ * 2);
    __bf16* Kh    = (__bf16*)alloc((size_t)B_ * H_ * SKV_ * DK_ * 2);
    __bf16* Vt    = (__bf16*)alloc((size_t)B_ * H_ * DK_ * SKV_ * 2);
    __bf16* P     = (__bf16*)alloc((size_t)B_ * H_ * SQ_ * SKV_ * 2);
    __bf16* ctx   = (__bf16*)alloc((size_t)B_ * SQ_ * D_ * 2);
    float*  x     = (float*) alloc((size_t)B_ * SQ_ * D_ * 4);
    __bf16* xln   = (__bf16*)alloc((size_t)B_ * SQ_ * D_ * 2);
    __bf16* h1    = (__bf16*)alloc((size_t)B_ * SQ_ * DFF_ * 2);

    // 1) pre-norms
    ln_bf16_kernel<<<B_ * SKV_, 256, 0, stream>>>(memory, g_kv, b_kv, kv_ln, D_);
    ln_bf16_kernel<<<B_ * SQ_,  256, 0, stream>>>(q,      g_q,  b_q,  q_ln,  D_);

    // 2) weight convert + transpose
    wtrans_kernel<<<dim3(D_ / 32, D_ / 32),   256, 0, stream>>>(Wq, Wq_t, D_, D_);
    wtrans_kernel<<<dim3(D_ / 32, D_ / 32),   256, 0, stream>>>(Wk, Wk_t, D_, D_);
    wtrans_kernel<<<dim3(D_ / 32, D_ / 32),   256, 0, stream>>>(Wv, Wv_t, D_, D_);
    wtrans_kernel<<<dim3(D_ / 32, D_ / 32),   256, 0, stream>>>(Wo, Wo_t, D_, D_);
    wtrans_kernel<<<dim3(D_ / 32, DFF_ / 32), 256, 0, stream>>>(W1, W1_t, D_, DFF_);
    wtrans_kernel<<<dim3(DFF_ / 32, D_ / 32), 256, 0, stream>>>(W2, W2_t, DFF_, D_);

    // 3) QKV projections (scale 1/sqrt(DK) folded into Q); MT=4 -> 512 rows / block
    qkv_proj_kernel<4><<<dim3(B_ * SQ_  / 512, D_ / 64), 256, 0, stream>>>(
        q_ln,  Wq_t, bq, Qh, 8,  0, 0.125f);    // logS = log2(SQ_) = 8
    qkv_proj_kernel<4><<<dim3(B_ * SKV_ / 512, D_ / 64), 256, 0, stream>>>(
        kv_ln, Wk_t, bk, Kh, 10, 1, 1.0f);      // logS = log2(SKV_) = 10
    qkv_proj_kernel<4><<<dim3(B_ * SKV_ / 512, D_ / 64), 256, 0, stream>>>(
        kv_ln, Wv_t, bv, Vt, 10, 2, 1.0f);

    // 4) attention scores -> attn region of d_out, softmax in place, bf16 copy (MT=2)
    scores_kernel<2><<<dim3(SQ_ / 256, SKV_ / 64, B_ * H_), 256, 0, stream>>>(
        Qh, Kh, attn_f32);
    softmax_kernel<<<B_ * H_ * SQ_, 256, 0, stream>>>(attn_f32, P);

    // 5) ctx = P @ V, heads merged (MT=2)
    ctx_kernel<2><<<dim3(SQ_ / 256, 1, B_ * H_), 256, 0, stream>>>(P, Vt, ctx);

    // 6) x = ctx @ Wo + bo (f32)
    gemm_f32out_kernel<4><<<dim3(B_ * SQ_ / 512, D_ / 64), 256, 0, stream>>>(
        ctx, Wo_t, bo, nullptr, x, D_, D_);

    // 7) FFN: out = gelu(LN(x) @ W1 + b1) @ W2 + b2 + x
    ln_bf16_kernel<<<B_ * SQ_, 256, 0, stream>>>(x, g_ff, b_ff, xln, D_);
    gemm_gelu_kernel<4><<<dim3(B_ * SQ_ / 512, DFF_ / 64), 256, 0, stream>>>(
        xln, W1_t, b1, h1, D_, DFF_);
    gemm_f32out_kernel<4><<<dim3(B_ * SQ_ / 512, D_ / 64), 256, 0, stream>>>(
        h1, W2_t, b2, x, out_f32, DFF_, D_);
}
